// GAT_27676769256196
// MI455X (gfx1250) — compile-verified, matrix operand order checked by reading, not run on hardware
//
#include <hip/hip_runtime.h>
#include <hip/hip_bf16.h>
#include <stdint.h>
#include <stddef.h>

#define NWN 50000
#define NDN 5000
#define EWWE 800000
#define EWDE 500000
#define DIN 256
#define DH 128

typedef __attribute__((ext_vector_type(16))) __bf16 v16bf;
typedef __attribute__((ext_vector_type(8)))  float  v8f;

// ---------- order-preserving float<->uint encoding for atomic max ----------
__device__ __forceinline__ unsigned encf(float f) {
    unsigned u = __float_as_uint(f);
    return (u & 0x80000000u) ? ~u : (u | 0x80000000u);
}
__device__ __forceinline__ float decf(unsigned e) {
    unsigned u = (e & 0x80000000u) ? (e & 0x7FFFFFFFu) : ~e;
    return __uint_as_float(u);
}
#define ENC_NEG_INF 0x007FFFFFu  // encf(-inf)

// ---------- conversion kernels ----------
__global__ void conv_bf16(const float* __restrict__ X, __bf16* __restrict__ Y, int n) {
    int i = blockIdx.x * blockDim.x + threadIdx.x;
    if (i < n) Y[i] = (__bf16)X[i];
}

// W: [K, DH] f32 row-major  ->  Wt: [DH, K] bf16 row-major (N-major)
__global__ void conv_transpose_w(const float* __restrict__ W, __bf16* __restrict__ Wt, int K) {
    int i = blockIdx.x * blockDim.x + threadIdx.x;
    if (i >= K * DH) return;
    int k = i / DH, n = i % DH;
    Wt[(size_t)n * K + k] = (__bf16)W[i];
}

__global__ void zero_f(float* __restrict__ p, int n) {
    int i = blockIdx.x * blockDim.x + threadIdx.x;
    if (i < n) p[i] = 0.0f;
}

__global__ void init_mz(unsigned* __restrict__ m, float* __restrict__ z, int n) {
    int i = blockIdx.x * blockDim.x + threadIdx.x;
    if (i < n) { m[i] = ENC_NEG_INF; z[i] = 0.0f; }
}

// ---------- WMMA GEMM: F[M,128] = X[M,K](bf16) * Wt[128,K]^T(bf16), f32 acc ----------
// 4 waves/block, each wave computes a 16x128 strip via 8 v_wmma_f32_16x16x32_bf16 per K-step.
__global__ __launch_bounds__(128) void gemm_bf16_wmma(
    const __bf16* __restrict__ X, const __bf16* __restrict__ Wt,
    float* __restrict__ F, int M, int K)
{
    const int lane  = threadIdx.x & 31;
    const int wave  = threadIdx.x >> 5;
    const int mBase = (blockIdx.x * 4 + wave) * 16;
    if (mBase >= M) return;                 // wave-uniform exit (EXEC stays full for WMMA)

    const int half = lane >> 4;             // 0: K chunk [kk,kk+8)+[kk+16,kk+24) base 0; 1: base 8
    const int l16  = lane & 15;
    int mRow = mBase + l16;
    if (mRow >= M) mRow = M - 1;            // clamp loads; stores guarded below
    const __bf16* arow = X + (size_t)mRow * K;

    v8f acc[8] = {};

    for (int k = 0; k < K; k += 32) {
        const int kk = k + half * 8;
        union { v16bf v; uint4 u[2]; } a;
        a.u[0] = *(const uint4*)(arow + kk);
        a.u[1] = *(const uint4*)(arow + kk + 16);
#pragma unroll
        for (int t = 0; t < 8; ++t) {
            union { v16bf v; uint4 u[2]; } b;
            const __bf16* brow = Wt + (size_t)(t * 16 + l16) * K;
            b.u[0] = *(const uint4*)(brow + kk);
            b.u[1] = *(const uint4*)(brow + kk + 16);
            acc[t] = __builtin_amdgcn_wmma_f32_16x16x32_bf16(
                false, a.v, false, b.v, (short)0, acc[t], false, false);
        }
    }
    // C layout: VGPR j -> row mBase + j + half*8, col = t*16 + l16
#pragma unroll
    for (int t = 0; t < 8; ++t) {
#pragma unroll
        for (int j = 0; j < 8; ++j) {
            int r = mBase + j + half * 8;
            if (r < M) F[(size_t)r * DH + t * 16 + l16] = acc[t][j];
        }
    }
}

// ---------- el/er: per-node dot(F[n,:], a[:]) over DH=128, one wave per node ----------
__global__ __launch_bounds__(256) void node_dot(
    const float* __restrict__ F, const float* __restrict__ a,
    float* __restrict__ out, int N)
{
    int node = blockIdx.x * 8 + (threadIdx.x >> 5);
    if (node >= N) return;
    int lane = threadIdx.x & 31;
    float4 f = *(const float4*)(F + (size_t)node * DH + lane * 4);
    float4 w = *(const float4*)(a + lane * 4);
    float s = f.x * w.x + f.y * w.y + f.z * w.z + f.w * w.w;
#pragma unroll
    for (int off = 16; off > 0; off >>= 1) s += __shfl_down(s, off, 32);
    if (lane == 0) out[node] = s;
}

// ---------- edge pass 1: leaky-relu logit + segment max ----------
__global__ void edge_logit_max(
    const int* __restrict__ src, const int* __restrict__ dst,
    const float* __restrict__ el, const float* __restrict__ er,
    float* __restrict__ e, unsigned* __restrict__ m, int E)
{
    int i = blockIdx.x * blockDim.x + threadIdx.x;
    if (i >= E) return;
    float v = el[src[i]] + er[dst[i]];
    v = v > 0.0f ? v : 0.2f * v;
    e[i] = v;
    atomicMax(m + dst[i], encf(v));
}

// ---------- edge pass 2: exp(e - m[dst]) + segment sum ----------
__global__ void edge_exp_sum(
    const int* __restrict__ dst, float* __restrict__ e,
    const unsigned* __restrict__ m, float* __restrict__ z, int E)
{
    int i = blockIdx.x * blockDim.x + threadIdx.x;
    if (i >= E) return;
    int d = dst[i];
    float mf = decf(m[d]);              // finite: this dst has >=1 edge (this one)
    float a = __expf(e[i] - mf);
    e[i] = a;
    atomicAdd(z + d, a);
}

// ---------- edge pass 3: acc[dst] += Fs[src] * alpha, one wave per edge ----------
__global__ __launch_bounds__(256) void edge_aggregate(
    const int* __restrict__ src, const int* __restrict__ dst,
    const float* __restrict__ e, const float* __restrict__ z,
    const float* __restrict__ Fs, float* __restrict__ acc, int E)
{
    int edge = blockIdx.x * 8 + (threadIdx.x >> 5);
    if (edge >= E) return;
    int lane = threadIdx.x & 31;
    int s = src[edge], d = dst[edge];
    float alpha = e[edge] / fmaxf(z[d], 1e-16f);
    float4 f = *(const float4*)(Fs + (size_t)s * DH + lane * 4);
    float* ap = acc + (size_t)d * DH + lane * 4;
    atomicAdd(ap + 0, f.x * alpha);
    atomicAdd(ap + 1, f.y * alpha);
    atomicAdd(ap + 2, f.z * alpha);
    atomicAdd(ap + 3, f.w * alpha);
}

// ---------- layer finalize: relu(acc + sum of biases) -> bf16 or f32 ----------
__global__ void finalize_bf(const float* __restrict__ acc,
                            const float* b0, const float* b1, const float* b2,
                            __bf16* __restrict__ out, int N)
{
    int i = blockIdx.x * blockDim.x + threadIdx.x;
    if (i >= N * DH) return;
    int d = i & (DH - 1);
    float bias = (b0 ? b0[d] : 0.0f) + (b1 ? b1[d] : 0.0f) + (b2 ? b2[d] : 0.0f);
    out[i] = (__bf16)fmaxf(acc[i] + bias, 0.0f);
}

__global__ void finalize_f32(const float* __restrict__ acc,
                             const float* b0, const float* b1, const float* b2,
                             float* __restrict__ out, int N)
{
    int i = blockIdx.x * blockDim.x + threadIdx.x;
    if (i >= N * DH) return;
    int d = i & (DH - 1);
    float bias = (b0 ? b0[d] : 0.0f) + (b1 ? b1[d] : 0.0f) + (b2 ? b2[d] : 0.0f);
    out[i] = fmaxf(acc[i] + bias, 0.0f);
}

// ---------- final linear + sigmoid, one wave per node ----------
__global__ __launch_bounds__(256) void linear_sigmoid(
    const float* __restrict__ X, const float* __restrict__ w,
    const float* __restrict__ bias, float* __restrict__ out, int N)
{
    int node = blockIdx.x * 8 + (threadIdx.x >> 5);
    if (node >= N) return;
    int lane = threadIdx.x & 31;
    float4 x  = *(const float4*)(X + (size_t)node * DH + lane * 4);
    float4 ww = *(const float4*)(w + lane * 4);
    float s = x.x * ww.x + x.y * ww.y + x.z * ww.z + x.w * ww.w;
#pragma unroll
    for (int off = 16; off > 0; off >>= 1) s += __shfl_down(s, off, 32);
    if (lane == 0) out[node] = 1.0f / (1.0f + __expf(-(s + bias[0])));
}

// =====================================================================
extern "C" void kernel_launch(void* const* d_in, const int* in_sizes, int n_in,
                              void* d_out, int out_size, void* d_ws, size_t ws_size,
                              hipStream_t stream)
{
    const float* xw    = (const float*)d_in[0];
    const float* xd    = (const float*)d_in[1];
    const float* W1    = (const float*)d_in[2];
    const float* al1   = (const float*)d_in[3];
    const float* ar1   = (const float*)d_in[4];
    const float* b1    = (const float*)d_in[5];
    const float* W2    = (const float*)d_in[6];
    const float* al2   = (const float*)d_in[7];
    const float* ar2   = (const float*)d_in[8];
    const float* b2    = (const float*)d_in[9];
    const float* lin_w = (const float*)d_in[10];
    const float* lin_b = (const float*)d_in[11];
    const int* src_ww  = (const int*)d_in[12];
    const int* dst_ww  = (const int*)d_in[13];
    const int* src_wwr = (const int*)d_in[14];
    const int* dst_wwr = (const int*)d_in[15];
    const int* src_wd  = (const int*)d_in[16];
    const int* dst_wd  = (const int*)d_in[17];
    const int* src_wdr = (const int*)d_in[18];
    const int* dst_wdr = (const int*)d_in[19];

    // ---- carve workspace ----
    char* ws = (char*)d_ws;
    size_t off = 0;
    auto carve = [&](size_t bytes) -> void* {
        void* p = ws + off;
        off += (bytes + 255) & ~(size_t)255;
        return p;
    };
    __bf16*  xw_bf = (__bf16*)carve((size_t)NWN * DIN * 2);
    __bf16*  xd_bf = (__bf16*)carve((size_t)NDN * DIN * 2);
    __bf16*  Wt    = (__bf16*)carve((size_t)DH * DIN * 2);     // reused per relation
    float*   Fs    = (float*)carve((size_t)NWN * DH * 4);
    float*   Fd    = (float*)carve((size_t)NWN * DH * 4);
    float*   el    = (float*)carve((size_t)NWN * 4);
    float*   er    = (float*)carve((size_t)NWN * 4);
    unsigned* mb   = (unsigned*)carve((size_t)NWN * 4);
    float*   zb    = (float*)carve((size_t)NWN * 4);
    float*   ebuf  = (float*)carve((size_t)EWWE * 4);
    float*   accw  = (float*)carve((size_t)NWN * DH * 4);
    float*   accd  = (float*)carve((size_t)NDN * DH * 4);
    __bf16*  x1w   = (__bf16*)carve((size_t)NWN * DH * 2);
    __bf16*  x1d   = (__bf16*)carve((size_t)NDN * DH * 2);
    float*   x2w   = (float*)carve((size_t)NWN * DH * 4);
    float*   x2d   = (float*)carve((size_t)NDN * DH * 4);

    // ---- initial bf16 conversion of node features ----
    conv_bf16<<<(NWN * DIN + 255) / 256, 256, 0, stream>>>(xw, xw_bf, NWN * DIN);
    conv_bf16<<<(NDN * DIN + 255) / 256, 256, 0, stream>>>(xd, xd_bf, NDN * DIN);

    // ---- one GAT relation ----
    auto run_rel = [&](const __bf16* Xs, int Ns, const __bf16* Xd, int Nd,
                       const float* Wr, const float* alr, const float* arr,
                       const int* srcI, const int* dstI, int E, float* accT, int K) {
        conv_transpose_w<<<(K * DH + 255) / 256, 256, 0, stream>>>(Wr, Wt, K);
        gemm_bf16_wmma<<<(Ns + 63) / 64, 128, 0, stream>>>(Xs, Wt, Fs, Ns, K);
        const float* FdP = Fs;
        if (Xd != Xs) {
            gemm_bf16_wmma<<<(Nd + 63) / 64, 128, 0, stream>>>(Xd, Wt, Fd, Nd, K);
            FdP = Fd;
        }
        node_dot<<<(Ns + 7) / 8, 256, 0, stream>>>(Fs, alr, el, Ns);
        node_dot<<<(Nd + 7) / 8, 256, 0, stream>>>(FdP, arr, er, Nd);
        init_mz<<<(Nd + 255) / 256, 256, 0, stream>>>(mb, zb, Nd);
        edge_logit_max<<<(E + 255) / 256, 256, 0, stream>>>(srcI, dstI, el, er, ebuf, mb, E);
        edge_exp_sum<<<(E + 255) / 256, 256, 0, stream>>>(dstI, ebuf, mb, zb, E);
        edge_aggregate<<<(E + 7) / 8, 256, 0, stream>>>(srcI, dstI, ebuf, zb, Fs, accT, E);
    };

    // ================= layer 1 (K = 256) =================
    zero_f<<<(NWN * DH + 255) / 256, 256, 0, stream>>>(accw, NWN * DH);
    zero_f<<<(NDN * DH + 255) / 256, 256, 0, stream>>>(accd, NDN * DH);
    const size_t WS1 = (size_t)DIN * DH;
    run_rel(xw_bf, NWN, xw_bf, NWN, W1 + 0 * WS1, al1 + 0 * DH, ar1 + 0 * DH,
            src_ww, dst_ww, EWWE, accw, DIN);
    run_rel(xw_bf, NWN, xd_bf, NDN, W1 + 1 * WS1, al1 + 1 * DH, ar1 + 1 * DH,
            src_wd, dst_wd, EWDE, accd, DIN);
    run_rel(xw_bf, NWN, xw_bf, NWN, W1 + 2 * WS1, al1 + 2 * DH, ar1 + 2 * DH,
            src_wwr, dst_wwr, EWWE, accw, DIN);
    run_rel(xd_bf, NDN, xw_bf, NWN, W1 + 3 * WS1, al1 + 3 * DH, ar1 + 3 * DH,
            src_wdr, dst_wdr, EWDE, accw, DIN);
    finalize_bf<<<(NWN * DH + 255) / 256, 256, 0, stream>>>(
        accw, b1 + 0 * DH, b1 + 2 * DH, b1 + 3 * DH, x1w, NWN);
    finalize_bf<<<(NDN * DH + 255) / 256, 256, 0, stream>>>(
        accd, b1 + 1 * DH, nullptr, nullptr, x1d, NDN);

    // ================= layer 2 (K = 128) =================
    zero_f<<<(NWN * DH + 255) / 256, 256, 0, stream>>>(accw, NWN * DH);
    zero_f<<<(NDN * DH + 255) / 256, 256, 0, stream>>>(accd, NDN * DH);
    const size_t WS2 = (size_t)DH * DH;
    run_rel(x1w, NWN, x1w, NWN, W2 + 0 * WS2, al2 + 0 * DH, ar2 + 0 * DH,
            src_ww, dst_ww, EWWE, accw, DH);
    run_rel(x1w, NWN, x1d, NDN, W2 + 1 * WS2, al2 + 1 * DH, ar2 + 1 * DH,
            src_wd, dst_wd, EWDE, accd, DH);
    run_rel(x1w, NWN, x1w, NWN, W2 + 2 * WS2, al2 + 2 * DH, ar2 + 2 * DH,
            src_wwr, dst_wwr, EWWE, accw, DH);
    run_rel(x1d, NDN, x1w, NWN, W2 + 3 * WS2, al2 + 3 * DH, ar2 + 3 * DH,
            src_wdr, dst_wdr, EWDE, accw, DH);
    finalize_f32<<<(NWN * DH + 255) / 256, 256, 0, stream>>>(
        accw, b2 + 0 * DH, b2 + 2 * DH, b2 + 3 * DH, x2w, NWN);
    finalize_f32<<<(NDN * DH + 255) / 256, 256, 0, stream>>>(
        accd, b2 + 1 * DH, nullptr, nullptr, x2d, NDN);

    // ================= readout =================
    float* outp = (float*)d_out;
    linear_sigmoid<<<(NWN + 7) / 8, 256, 0, stream>>>(x2w, lin_w, lin_b, outp, NWN);
    linear_sigmoid<<<(NDN + 7) / 8, 256, 0, stream>>>(x2d, lin_w, lin_b, outp + NWN, NDN);
}
